// PoseGNN_59047210385938
// MI455X (gfx1250) — matrix operand order repaired and checked by reading, not compiled, not run on hardware
//
#include <hip/hip_runtime.h>
#include <hip/hip_bf16.h>
#include <math.h>

typedef __attribute__((ext_vector_type(16))) __bf16 v16bf;
typedef __attribute__((ext_vector_type(8)))  __bf16 v8bf;
typedef __attribute__((ext_vector_type(8)))  float  v8f;
typedef int v4i __attribute__((vector_size(16)));

#define AS1 __attribute__((address_space(1)))
#define AS3 __attribute__((address_space(3)))

#define FDIM 128
#define LDW  136   // padded LDS row stride (elements); 136*2B = 272B, 16B aligned

#if __has_builtin(__builtin_amdgcn_global_load_async_to_lds_b128)
#define HAVE_ASYNC_LDS 1
#endif

// ---------------------------------------------------------------------------
// degree / normalization
// ---------------------------------------------------------------------------
__global__ __launch_bounds__(256) void init_deg(float* deg, int N) {
    int i = blockIdx.x * 256 + threadIdx.x;
    if (i < N) deg[i] = 1.0f;             // self-loop
}

__global__ __launch_bounds__(256) void count_deg(float* deg, const long long* dst, int E) {
    int e = blockIdx.x * 256 + threadIdx.x;
    if (e < E) unsafeAtomicAdd(&deg[dst[e]], 1.0f);
}

__global__ __launch_bounds__(256) void rsqrt_deg(float* dis, int N) {
    int i = blockIdx.x * 256 + threadIdx.x;
    if (i < N) {
        float d = dis[i];
        dis[i] = d > 0.0f ? rsqrtf(d) : 0.0f;
    }
}

// ---------------------------------------------------------------------------
// GEMM: out[M x 128] = A[M x 128] (fp32, cvt->bf16) @ W[128 x 128] (fp32, cvt->bf16)
// fp32 accumulate via v_wmma_f32_16x16x32_bf16. One 16-row strip per wave.
// W staged into LDS via async global->LDS copies (ASYNCcnt), then cvt+transpose.
// ---------------------------------------------------------------------------
__global__ __launch_bounds__(256) void gemm_bf16_wmma(
    const float* __restrict__ A, const float* __restrict__ W,
    float* __restrict__ out, int M)
{
    __shared__ float  Wstage[32 * FDIM];  // 16 KB staging (1/4 of W per pass)
    __shared__ __bf16 Wt[FDIM * LDW];     // transposed bf16: Wt[col*LDW + row]

    int tid = threadIdx.x;

    // stage W through LDS in 4 passes of 32 rows, convert + transpose to bf16
    for (int p = 0; p < 4; ++p) {
        const float* wsrc = W + p * (32 * FDIM);
#ifdef HAVE_ASYNC_LDS
#pragma unroll
        for (int it = 0; it < 4; ++it) {
            int v4 = it * 256 + tid;      // float4 index, 0..1023
            __builtin_amdgcn_global_load_async_to_lds_b128(
                (AS1 v4i*)(uintptr_t)(wsrc + v4 * 4),
                (AS3 v4i*)(unsigned)(uintptr_t)(&Wstage[v4 * 4]),
                0, 0);
        }
#if __has_builtin(__builtin_amdgcn_s_wait_asynccnt)
        __builtin_amdgcn_s_wait_asynccnt(0);
#else
        asm volatile("s_wait_asynccnt 0" ::: "memory");
#endif
#else
        for (int v4 = tid; v4 < 1024; v4 += 256)
            ((float4*)Wstage)[v4] = ((const float4*)wsrc)[v4];
#endif
        __syncthreads();
        for (int i = tid; i < 32 * FDIM; i += 256) {
            int rl = i >> 7, c = i & 127;
            Wt[c * LDW + (p * 32 + rl)] = (__bf16)Wstage[i];
        }
        __syncthreads();
    }

    int wave = tid >> 5;
    int lane = tid & 31;
    int m = lane & 15;        // row-in-tile (A,M) / col-in-tile (B,D,N)
    int g = lane >> 4;        // lane-group select

    int strips = (M + 15) >> 4;
    int strip = blockIdx.x * 8 + wave;
    if (strip >= strips) return;          // wave-uniform: EXEC stays all-ones

    int rowBase = strip * 16;
    int row = rowBase + m;
    int rc = row < M ? row : M - 1;
    const float* ap = A + (size_t)rc * FDIM;

    // Build 4 A fragments (16x32 bf16 each) per the CDNA5 16-bit A layout:
    //   a[0..7]  = A[m][kb*32 + g*8 + j]
    //   a[8..15] = A[m][kb*32 + 16 + g*8 + j]
    v16bf afrag[4];
#pragma unroll
    for (int kb = 0; kb < 4; ++kb) {
        int base = kb * 32 + g * 8;
#pragma unroll
        for (int j = 0; j < 8; ++j) {
            afrag[kb][j]     = (__bf16)ap[base + j];
            afrag[kb][j + 8] = (__bf16)ap[base + 16 + j];
        }
    }

    bool fullStrip = (rowBase + 16) <= M;  // wave-uniform guard

    for (int nt = 0; nt < 8; ++nt) {
        v8f acc = {};
#pragma unroll
        for (int kb = 0; kb < 4; ++kb) {
            // B fragment: b[i] = W[kb*32 + g*8 + i][nt*16 + m], b[i+8] = W[kb*32 + 16 + g*8 + i][nt*16 + m]
            const __bf16* bp = &Wt[(nt * 16 + m) * LDW + kb * 32 + g * 8];
            v8bf lo = *(const v8bf*)(bp);
            v8bf hi = *(const v8bf*)(bp + 16);
            v16bf bfrag;
#pragma unroll
            for (int i = 0; i < 8; ++i) { bfrag[i] = lo[i]; bfrag[i + 8] = hi[i]; }
            acc = __builtin_amdgcn_wmma_f32_16x16x32_bf16(
                false, afrag[kb], false, bfrag, (short)0, acc, false, false);
        }
        // D layout: lane holds D[rowBase + r + 8*g][nt*16 + m] in acc[r]
        float* ob = out + ((size_t)(rowBase + 8 * g) * FDIM) + nt * 16 + m;
        if (fullStrip) {
#pragma unroll
            for (int r = 0; r < 8; ++r) ob[(size_t)r * FDIM] = acc[r];
        } else {
#pragma unroll
            for (int r = 0; r < 8; ++r) {
                int orow = rowBase + r + 8 * g;
                if (orow < M) out[(size_t)orow * FDIM + nt * 16 + m] = acc[r];
            }
        }
    }
}

// ---------------------------------------------------------------------------
// aggregation
// ---------------------------------------------------------------------------
__global__ __launch_bounds__(256) void init_ag(
    float* AG, const float* __restrict__ HW, const float* __restrict__ dis, int N)
{
    int i = blockIdx.x * 256 + threadIdx.x;
    if (i < N * FDIM) {
        int n = i >> 7;
        float s = dis[n];
        AG[i] = s * s * HW[i];            // self-loop term
    }
}

__global__ __launch_bounds__(256) void scatter_edges(
    float* AG, const float* __restrict__ HW, const float* __restrict__ dis,
    const long long* __restrict__ src, const long long* __restrict__ dst, int E)
{
    int t = blockIdx.x * 256 + threadIdx.x;
    int e = t >> 7;                       // 128 lanes per edge: coalesced 512B row
    int j = t & 127;
    if (e >= E) return;
    long long s = src[e], d = dst[e];
    float coef = dis[s] * dis[d];
    unsafeAtomicAdd(&AG[(size_t)d * FDIM + j], HW[(size_t)s * FDIM + j] * coef);
}

__global__ __launch_bounds__(256) void finalize_relu(
    float* X2, const float* __restrict__ AG, const float* __restrict__ b, int N)
{
    int i = blockIdx.x * 256 + threadIdx.x;
    if (i < N * FDIM) {
        float v = AG[i] + b[i & 127];
        X2[i] = v > 0.0f ? v : 0.0f;
    }
}

// ---------------------------------------------------------------------------
// readout
// ---------------------------------------------------------------------------
__global__ __launch_bounds__(128) void zero_vec(float* g) {
    g[threadIdx.x] = 0.0f;
}

__global__ __launch_bounds__(128) void col_sum(
    const float* __restrict__ H, float* gsum, int N)
{
    int j = threadIdx.x;                  // column 0..127
    int r0 = blockIdx.x * 256;
    int rend = r0 + 256; if (rend > N) rend = N;
    float s = 0.0f;
    for (int r = r0; r < rend; ++r) s += H[(size_t)r * FDIM + j];
    unsafeAtomicAdd(&gsum[j], s);
}

__global__ __launch_bounds__(32) void head_kernel(
    const float* __restrict__ gsum,
    const float* __restrict__ Wf1, const float* __restrict__ bf1,
    const float* __restrict__ Wf2, const float* __restrict__ bf2,
    float* out, int N, int O1, int O2)
{
    __shared__ float gm[FDIM];
    __shared__ float logits[32];
    int l = threadIdx.x;
    float invN = 1.0f / (float)N;
    for (int j = l; j < FDIM; j += 32) gm[j] = gsum[j] * invN;
    __syncthreads();

    if (l < O1) {
        float acc = bf1[l];
        for (int j = 0; j < FDIM; ++j) acc += gm[j] * Wf1[j * O1 + l];
        logits[l] = acc;
    }
    if (l < O2) {
        float acc = bf2[l];
        for (int j = 0; j < FDIM; ++j) acc += gm[j] * Wf2[j * O2 + l];
        logits[16 + l] = acc;
    }
    __syncthreads();
    if (l == 0) {
        float mx = logits[0];
        for (int o = 1; o < O1; ++o) mx = fmaxf(mx, logits[o]);
        float ss = 0.0f;
        for (int o = 0; o < O1; ++o) { float e = expf(logits[o] - mx); out[o] = e; ss += e; }
        float inv = 1.0f / ss;
        for (int o = 0; o < O1; ++o) out[o] *= inv;

        mx = logits[16];
        for (int o = 1; o < O2; ++o) mx = fmaxf(mx, logits[16 + o]);
        ss = 0.0f;
        for (int o = 0; o < O2; ++o) { float e = expf(logits[16 + o] - mx); out[O1 + o] = e; ss += e; }
        inv = 1.0f / ss;
        for (int o = 0; o < O2; ++o) out[O1 + o] *= inv;
    }
}

// ---------------------------------------------------------------------------
extern "C" void kernel_launch(void* const* d_in, const int* in_sizes, int n_in,
                              void* d_out, int out_size, void* d_ws, size_t ws_size,
                              hipStream_t stream) {
    const float*     x   = (const float*)d_in[0];
    const long long* ei  = (const long long*)d_in[1];
    const float*     W1  = (const float*)d_in[2];
    const float*     b1  = (const float*)d_in[3];
    const float*     W2  = (const float*)d_in[4];
    const float*     b2  = (const float*)d_in[5];
    const float*     Wf1 = (const float*)d_in[6];
    const float*     bf1 = (const float*)d_in[7];
    const float*     Wf2 = (const float*)d_in[8];
    const float*     bf2 = (const float*)d_in[9];

    int N  = in_sizes[0] / FDIM;
    int E  = in_sizes[1] / 2;
    int O1 = in_sizes[7];
    int O2 = in_sizes[9];
    float* out = (float*)d_out;

    char* ws = (char*)d_ws;
    size_t o0  = ((size_t)N * 4 + 255) & ~(size_t)255;
    size_t NH  = (size_t)N * FDIM * 4;
    float* dis  = (float*)ws;                 // deg -> dis (in place)
    float* HW   = (float*)(ws + o0);          // X @ W
    float* AG   = (float*)(ws + o0 + NH);     // aggregated messages
    float* XB   = (float*)(ws + o0 + 2 * NH); // layer activations
    float* gsum = (float*)(ws + o0 + 3 * NH); // column sums (128)

    const long long* src = ei;
    const long long* dst = ei + E;

    int strips = (N + 15) / 16;
    int gemmBlocks = (strips + 7) / 8;
    int nEl = N * FDIM;

    // normalization
    init_deg  <<<(N + 255) / 256, 256, 0, stream>>>(dis, N);
    count_deg <<<(E + 255) / 256, 256, 0, stream>>>(dis, dst, E);
    rsqrt_deg <<<(N + 255) / 256, 256, 0, stream>>>(dis, N);

    // layer 1
    gemm_bf16_wmma<<<gemmBlocks, 256, 0, stream>>>(x, W1, HW, N);
    init_ag       <<<(nEl + 255) / 256, 256, 0, stream>>>(AG, HW, dis, N);
    scatter_edges <<<(unsigned)(((size_t)E * FDIM + 255) / 256), 256, 0, stream>>>(AG, HW, dis, src, dst, E);
    finalize_relu <<<(nEl + 255) / 256, 256, 0, stream>>>(XB, AG, b1, N);

    // layer 2
    gemm_bf16_wmma<<<gemmBlocks, 256, 0, stream>>>(XB, W2, HW, N);
    init_ag       <<<(nEl + 255) / 256, 256, 0, stream>>>(AG, HW, dis, N);
    scatter_edges <<<(unsigned)(((size_t)E * FDIM + 255) / 256), 256, 0, stream>>>(AG, HW, dis, src, dst, E);
    finalize_relu <<<(nEl + 255) / 256, 256, 0, stream>>>(XB, AG, b2, N);

    // readout + heads
    zero_vec   <<<1, 128, 0, stream>>>(gsum);
    col_sum    <<<(N + 255) / 256, 128, 0, stream>>>(XB, gsum, N);
    head_kernel<<<1, 32, 0, stream>>>(gsum, Wf1, bf1, Wf2, bf2, out, N, O1, O2);
}
